// LSTMAutoregressiveDecoder_81939386073793
// MI455X (gfx1250) — matrix operand order repaired and checked by reading, not compiled, not run on hardware
//
#include <hip/hip_runtime.h>
#include <hip/hip_bf16.h>

typedef _Float16 v16h __attribute__((ext_vector_type(16)));
typedef float    v8f  __attribute__((ext_vector_type(8)));

#define BATCH   512
#define TSEQ    64
#define FIN     32
#define HID     512
#define GATES   2048   // 4*HID
#define OUTD    32
#define KTILES  17     // (FIN+HID)/32 augmented K tiles
#define NTILES  128    // GATES/16
#define NTHREADS 1024  // 32 waves: one 16-col h-tile per wave
#define STEPS_TOTAL 127  // 64 warmup + 63 decode cell steps
#define GSTRIDE (32 * KTILES * 32)   // v16h elements between gate blocks

// ---------------------------------------------------------------------------
// Fragment maps (wave32, 16-bit operands), per CDNA5 ISA 7.12.2:
//  A (16x32 MxK):  lane l, half q:  M = l%16,
//                  K = (q%8) + 8*( (l>=16) + 2*(q/8) )
//  B (32x16 KxN):  lane l, half q:  N = l%16,  K = (q%8) + 8*( (l>=16) + 2*(q/8) )
//  C/D (16x16):    lane l, reg r:   M = r + 8*(l/16), N = l%16
// ---------------------------------------------------------------------------

// Pack W = [Wx; Wh] (544 x 2048, fp32) into f16 B-fragment order.
__global__ void pack_w_kernel(const float* __restrict__ Wx,
                              const float* __restrict__ Wh,
                              _Float16* __restrict__ Wp) {
  int idx = blockIdx.x * blockDim.x + threadIdx.x;
  const int total = NTILES * KTILES * 512;
  if (idx >= total) return;
  int ntile = idx / (KTILES * 512);
  int rem   = idx % (KTILES * 512);
  int kt    = rem / 512;
  int e     = rem % 512;
  int lane  = e / 16;
  int q     = e % 16;
  int K = kt * 32 + (q & 7) + 8 * (lane / 16) + 16 * (q / 8);
  int n = ntile * 16 + (lane % 16);
  float v = (K < FIN) ? Wx[K * GATES + n] : Wh[(K - FIN) * GATES + n];
  Wp[idx] = (_Float16)v;
}

// Pack Wd (512 x 32, fp32) into f16 B-fragment order (16 k-tiles, 2 n-tiles).
__global__ void pack_wd_kernel(const float* __restrict__ Wd,
                               _Float16* __restrict__ Wdp) {
  int idx = blockIdx.x * blockDim.x + threadIdx.x;
  const int total = 2 * 16 * 512;
  if (idx >= total) return;
  int ntile = idx / (16 * 512);
  int kt    = (idx / 512) % 16;
  int e     = idx % 512;
  int lane  = e / 16;
  int q     = e % 16;
  int K = kt * 32 + (q & 7) + 8 * (lane / 16) + 16 * (q / 8);
  int n = ntile * 16 + (lane % 16);
  Wdp[idx] = (_Float16)Wd[K * OUTD + n];
}

// fast sigmoid / tanh: 2 TRANS ops each (v_exp_f32 + v_rcp_f32);
// co-executes with XDL WMMA (WMMA is TRANS-class for scheduling).
__device__ __forceinline__ float sigf(float v) {
  return __builtin_amdgcn_rcpf(1.0f + __expf(-v));
}
__device__ __forceinline__ float tanhfast(float v) {
  return 1.0f - 2.0f * __builtin_amdgcn_rcpf(1.0f + __expf(2.0f * v));
}

// Opaque zero: defeats LICM (the weight addresses become loop-variant, so
// loads are re-streamed from L2 each step instead of hoisted + spilled)
// WITHOUT laundering the pointer itself — address-space inference keeps the
// base provably global, so loads lower to global_load_b128 (LOADcnt only),
// not flat_load_b128 (LOADcnt+DScnt).
__device__ __forceinline__ int opaque_zero() {
  int v = 0;
  asm volatile("" : "+v"(v));
  return v;
}

// Persistent LSTM kernel: one block per 16-row batch tile, 32 waves.
// State: c in VGPRs (C-fragment layout), h in LDS (A-fragment layout, f16).
__global__ __launch_bounds__(NTHREADS)
void lstm_main(const float* __restrict__ x,
               const float* __restrict__ bg,
               const float* __restrict__ bd,
               const _Float16* __restrict__ Wp16,
               const _Float16* __restrict__ Wdp16,
               float* __restrict__ out) {
  // Augmented activation [in32 | h512] as 17 A-fragments (f16), 32B aligned
  __shared__ __align__(32) _Float16 afrag[KTILES * 512];

  const int tid   = threadIdx.x;
  const int wave  = tid >> 5;    // 0..31: h-tile owned by this wave
  const int lane  = tid & 31;
  const int btile = blockIdx.x;
  const int lhi   = lane >> 4;   // which half-wave (C-layout M offset)
  const int ln    = lane & 15;   // N column within a 16-wide tile

  // per-wave weight base; gate blocks at constant 557056-byte offsets,
  // k-tiles at constant 1024-byte offsets.
  const v16h* const wbase =
      (const v16h*)Wp16 + (size_t)wave * (KTILES * 32) + lane;
  const v16h* const dbase =
      (const v16h*)Wdp16 + (size_t)(wave & 1) * (16 * 32) + lane;

  // zero-init augmented fragments (h0 = 0)
  for (int i = tid; i < KTILES * 512; i += NTHREADS)
    afrag[i] = (_Float16)0.0f;

  // per-lane gate biases: gate col = gg*512 + wave*16 + ln (constant)
  float bias[4];
#pragma unroll
  for (int gg = 0; gg < 4; ++gg)
    bias[gg] = bg[gg * HID + wave * 16 + ln];
  float bdv = (wave < 2) ? bd[wave * 16 + ln] : 0.0f;

  v8f cst;
#pragma unroll
  for (int r = 0; r < 8; ++r) cst[r] = 0.0f;
  float hnew[8];

  // precomputed scatter coords for h_new -> A-fragment (k = 32 + hcol)
  const int hcol_s = wave * 16 + ln;
  const int k_s    = FIN + hcol_s;
  const int kt_s   = k_s >> 5;
  const int kk_s   = k_s & 31;
  const int q_s    = (kk_s & 7) + 8 * (kk_s >> 4);
  const int la_s   = 16 * ((kk_s >> 3) & 1);   // + m
  _Float16* const hdst = &afrag[kt_s * 512 + q_s];

  for (int it = 0; it < STEPS_TOTAL; ++it) {
    if (it < TSEQ && tid < 512) {
      // stage x[:, it, :] (16x32 fp32) into A-fragment 0 as f16
      int m  = tid >> 5;
      int kk = tid & 31;
      float v = x[((size_t)(btile * 16 + m) * TSEQ + it) * FIN + kk];
      int bit = (kk >> 3) & 1;
      int g   = kk >> 4;
      int q   = (kk & 7) + 8 * g;
      afrag[(m + 16 * bit) * 16 + q] = (_Float16)v;
    }
    __syncthreads();

    // ---- gate GEMM: z = [in|h] @ W ; wave owns h-tile `wave`, 4 gates ----
    {
      // loop-variant (but always-zero) offset: re-stream weights from L2
      const v16h* wp = wbase + opaque_zero();
      v8f acc0 = {}, acc1 = {}, acc2 = {}, acc3 = {};
#pragma unroll
      for (int kt = 0; kt < KTILES; ++kt) {
        v16h a  = *(const v16h*)&afrag[kt * 512 + lane * 16];
        v16h b0 = wp[kt * 32];
        v16h b1 = wp[kt * 32 + GSTRIDE];
        v16h b2 = wp[kt * 32 + 2 * GSTRIDE];
        v16h b3 = wp[kt * 32 + 3 * GSTRIDE];
        acc0 = __builtin_amdgcn_wmma_f32_16x16x32_f16(false, a, false, b0, (short)0, acc0, false, false);
        acc1 = __builtin_amdgcn_wmma_f32_16x16x32_f16(false, a, false, b1, (short)0, acc1, false, false);
        acc2 = __builtin_amdgcn_wmma_f32_16x16x32_f16(false, a, false, b2, (short)0, acc2, false, false);
        acc3 = __builtin_amdgcn_wmma_f32_16x16x32_f16(false, a, false, b3, (short)0, acc3, false, false);
      }
      // elementwise cell update (fp32), c kept in registers
#pragma unroll
      for (int r = 0; r < 8; ++r) {
        float ig = sigf(acc0[r] + bias[0]);
        float fg = sigf(acc1[r] + bias[1]);
        float gv = tanhfast(acc2[r] + bias[2]);
        float og = sigf(acc3[r] + bias[3]);
        float cn = fg * cst[r] + ig * gv;
        cst[r]   = cn;
        hnew[r]  = og * tanhfast(cn);
      }
    }
    __syncthreads();  // all reads of old afrag done before overwriting h

    // ---- scatter h_new (C-layout) into A-fragment layout ----
#pragma unroll
    for (int r = 0; r < 8; ++r) {
      int m = r + 8 * lhi;
      hdst[(m + la_s) * 16] = (_Float16)hnew[r];
    }

    // ---- dense head p = h @ Wd + bd (decode region), waves 0..1 ----------
    if (it >= TSEQ - 1) {
      __syncthreads();  // h fragments visible
      if (wave < 2) {
        const v16h* dp = dbase + opaque_zero();
        v8f p = {};
#pragma unroll
        for (int kt = 0; kt < 16; ++kt) {
          v16h a  = *(const v16h*)&afrag[(kt + 1) * 512 + lane * 16];
          v16h bw = dp[kt * 32];
          p = __builtin_amdgcn_wmma_f32_16x16x32_f16(false, a, false, bw, (short)0, p, false, false);
        }
        int col = wave * 16 + ln;
        int s   = it - (TSEQ - 1);
        int bit = (col >> 3) & 1;
        int g   = col >> 4;
        int q   = (col & 7) + 8 * g;
#pragma unroll
        for (int r = 0; r < 8; ++r) {
          int m = r + 8 * lhi;
          float pv = p[r] + bdv;
          out[((size_t)(btile * 16 + m) * TSEQ + s) * OUTD + col] = pv;
          // feed prediction back as next-step input (fragment 0)
          afrag[(m + 16 * bit) * 16 + q] = (_Float16)pv;
        }
      }
    }
    // next-iteration __syncthreads() makes frag0 writes visible before GEMM
  }
}

extern "C" void kernel_launch(void* const* d_in, const int* in_sizes, int n_in,
                              void* d_out, int out_size, void* d_ws, size_t ws_size,
                              hipStream_t stream) {
  const float* x  = (const float*)d_in[0];
  const float* Wx = (const float*)d_in[1];
  const float* Wh = (const float*)d_in[2];
  const float* bg = (const float*)d_in[3];
  const float* Wd = (const float*)d_in[4];
  const float* bd = (const float*)d_in[5];

  _Float16* Wp  = (_Float16*)d_ws;                       // 544*2048 halves
  _Float16* Wdp = Wp + (size_t)NTILES * KTILES * 512;    // 512*32 halves

  const int totW = NTILES * KTILES * 512;
  pack_w_kernel<<<(totW + 255) / 256, 256, 0, stream>>>(Wx, Wh, Wp);
  const int totWd = 2 * 16 * 512;
  pack_wd_kernel<<<(totWd + 255) / 256, 256, 0, stream>>>(Wd, Wdp);

  lstm_main<<<BATCH / 16, NTHREADS, 0, stream>>>(x, bg, bd, Wp, Wdp,
                                                 (float*)d_out);
}
